// WinAttn_43327630082427
// MI455X (gfx1250) — matrix operand order repaired and checked by reading, not compiled, not run on hardware
//
#include <hip/hip_runtime.h>

typedef _Float16 h16;
typedef __attribute__((ext_vector_type(16))) _Float16 v16h;
typedef __attribute__((ext_vector_type(8)))  _Float16 v8h;
typedef __attribute__((ext_vector_type(4)))  _Float16 v4h;
typedef __attribute__((ext_vector_type(8)))  float    v8f;
typedef __attribute__((ext_vector_type(4)))  float    v4f;

#define PX 136   // row pitch (halves) for 64x128 token-major tiles (16B aligned, odd*8 banks)
#define PV 72    // row pitch (halves) for Vt[c][m] and P[m][m']

#define WMMA(a,b,c) __builtin_amdgcn_wmma_f32_16x16x32_f16(false,(a),false,(b),(short)0,(c),false,false)

static __device__ __forceinline__ v16h comb16(v8h lo, v8h hi){
  v16h r;
#pragma unroll
  for(int i=0;i<8;++i){ r[i]=lo[i]; r[i+8]=hi[i]; }
  return r;
}

// A fragment 16x32 f16: rows m_base..m_base+15 (lane&15), K block at k0.
// ISA layout: lanes g=0: VGPR0-3 K=k0+0..7, VGPR4-7 K=k0+16..23 ; g=1: +8.
static __device__ __forceinline__ v16h loadA(const h16* __restrict__ base, int pitch,
                                             int m_base, int k0, int lane){
  int m = m_base + (lane & 15);
  int g = lane >> 4;
  const h16* p = base + m*pitch + k0 + 8*g;
  return comb16(*(const v8h*)p, *(const v8h*)(p+16));
}

// B fragment 32x16 f16 from row-major [n][k] buffer: lane holds column n,
// K = k0 + 8*g + 0..7, then k0 + 16 + 8*g + 0..7 (matches loadA).
static __device__ __forceinline__ v16h loadB(const h16* __restrict__ base, int pitch,
                                             int n_base, int k0, int lane){
  int n = n_base + (lane & 15);
  int g = lane >> 4;
  const h16* p = base + n*pitch + k0 + 8*g;
  return comb16(*(const v8h*)p, *(const v8h*)(p+16));
}

// ---- prep: transpose weights to [out][in] and convert fp32 -> f16 (L2-resident) ----
__global__ __launch_bounds__(256) void prep_weights(
    const float* __restrict__ Wq, const float* __restrict__ Wkv,
    const float* __restrict__ Wproj,
    h16* __restrict__ WqT, h16* __restrict__ WkvT, h16* __restrict__ WpT)
{
  int i  = blockIdx.x*256 + threadIdx.x;   // 0..32767
  int n  = i >> 7;                          // 0..255
  int k  = i & 127;
  WkvT[n*128 + k] = (h16)Wkv[k*256 + n];    // Wkv [128][256] -> [256][128]
  if (n < 128) {
    WqT[n*128 + k] = (h16)Wq[k*128 + n];
    WpT[n*128 + k] = (h16)Wproj[k*128 + n];
  }
}

// ---- fused window attention: one workgroup (4 wave32) per 8x8 window ----
__global__ __launch_bounds__(128) void winattn_fused(
    const float* __restrict__ x, const float* __restrict__ bq,
    const float* __restrict__ bkv, const float* __restrict__ rel_bias,
    const float* __restrict__ bproj,
    const h16* __restrict__ WqT, const h16* __restrict__ WkvT,
    const h16* __restrict__ WpT, float* __restrict__ out)
{
  extern __shared__ h16 smem[];
  h16* s_xq = smem;                 // 64*PX : X tile, overwritten in place by Q*scale
  h16* s_k  = s_xq + 64*PX;         // 64*PX : K, token-major [m][c]
  h16* s_vt = s_k  + 64*PX;         // 128*PV: V transposed [c][m]
  h16* s_p  = s_vt + 128*PV;        // 64*PV : softmax probs (per head)
  h16* s_o  = s_p  + 64*PV;         // 64*PX : attention output, token-major

  const int tid  = threadIdx.x;
  const int lane = tid & 31;
  const int wave = tid >> 5;
  const int ln   = lane & 15;
  const int g    = lane >> 4;
  const int m_base = wave * 16;     // this wave's 16 query rows

  const int b   = blockIdx.x >> 10;
  const int hy1 = (blockIdx.x >> 5) & 31;
  const int wx1 = blockIdx.x & 31;

  // ---- stage 1: stage window (64 tokens x 128 ch) into LDS as f16 ----
  // "faithful reshape" of the BCHW buffer => channels are the fastest dim:
  // elem(token T, c) = x[b*8388608 + T*128 + c], T = (hy1*8+iy)*256 + wx1*8+ix
  {
    const float4* xg4 = (const float4*)x;
    const int base4 = b * 2097152;           // 8388608/4
#pragma unroll
    for (int it = 0; it < 16; ++it) {
      int f  = tid + 128*it;                 // 0..2047 float4 slots
      int t  = f >> 5;                       // token 0..63
      int cq = f & 31;                       // float4 within channels
      int T  = (hy1*8 + (t>>3))*256 + wx1*8 + (t&7);
      float4 v = xg4[base4 + T*32 + cq];
      v4h hv; hv[0]=(h16)v.x; hv[1]=(h16)v.y; hv[2]=(h16)v.z; hv[3]=(h16)v.w;
      *(v4h*)(s_xq + t*PX + cq*4) = hv;
    }
  }
  __syncthreads();

  // ---- stage 2: QKV projections (wave owns its 16 rows) ----
  v16h xa[4];
#pragma unroll
  for (int ks = 0; ks < 4; ++ks) xa[ks] = loadA(s_xq, PX, m_base, 32*ks, lane);

  const float scale = 0.17677669529663687f;  // 32^-0.5 folded into stored Q

  // Q -> overwrite own rows of s_xq (X tile already fully in xa registers)
#pragma unroll
  for (int nt = 0; nt < 8; ++nt) {
    v8f acc = {};
#pragma unroll
    for (int ks = 0; ks < 4; ++ks)
      acc = WMMA(xa[ks], loadB(WqT, 128, nt*16, 32*ks, lane), acc);
    int c = nt*16 + ln;
    float bias = bq[c];
#pragma unroll
    for (int r = 0; r < 8; ++r)
      s_xq[(m_base + 8*g + r)*PX + c] = (h16)((acc[r] + bias) * scale);
  }
  // KV: output cols 0..127 = K (token-major), 128..255 = V (store transposed)
#pragma unroll
  for (int nt = 0; nt < 16; ++nt) {
    v8f acc = {};
#pragma unroll
    for (int ks = 0; ks < 4; ++ks)
      acc = WMMA(xa[ks], loadB(WkvT, 128, nt*16, 32*ks, lane), acc);
    int c = nt*16 + ln;
    float bias = bkv[c];
    if (nt < 8) {
#pragma unroll
      for (int r = 0; r < 8; ++r)
        s_k[(m_base + 8*g + r)*PX + c] = (h16)(acc[r] + bias);
    } else {
      int cv = c - 128;
      v8h pk;
#pragma unroll
      for (int r = 0; r < 8; ++r) pk[r] = (h16)(acc[r] + bias);
      *(v8h*)(s_vt + cv*PV + m_base + 8*g) = pk;   // contiguous m across r
    }
  }
  __syncthreads();

  // ---- stage 3: per-head attention. S = Q_h * K_h^T (K-dim = 32, 1 WMMA/tile)
#pragma unroll 1
  for (int h = 0; h < 4; ++h) {
    v16h aq = loadA(s_xq, PX, m_base, h*32, lane);
    v8f s[4];
#pragma unroll
    for (int j = 0; j < 4; ++j) {
      v8f z = {};
      s[j] = WMMA(aq, loadB(s_k, PX, j*16, h*32, lane), z);
    }
    // rel-pos bias + row softmax in C-tile layout: row m = m_base+8g+r lives in
    // VGPR r across the 16 lanes of half-wave g (4 tiles cover n=0..63).
    const int iy = (m_base + 8*g) >> 3;      // m%8 == r (m_base+8g is mult of 8)
#pragma unroll
    for (int r = 0; r < 8; ++r) {
      float vmax = -3.0e38f;
#pragma unroll
      for (int j = 0; j < 4; ++j) {
        int mp  = j*16 + ln;
        int idx = (iy - (mp>>3) + 7)*15 + (r - (mp&7) + 7);
        float sv = s[j][r] + rel_bias[idx*4 + h];
        s[j][r] = sv;
        vmax = fmaxf(vmax, sv);
      }
#pragma unroll
      for (int sh = 1; sh < 16; sh <<= 1)
        vmax = fmaxf(vmax, __shfl_xor(vmax, sh, 32));
      float sum = 0.f;
#pragma unroll
      for (int j = 0; j < 4; ++j) {
        float e = __expf(s[j][r] - vmax);
        s[j][r] = e; sum += e;
      }
#pragma unroll
      for (int sh = 1; sh < 16; sh <<= 1)
        sum += __shfl_xor(sum, sh, 32);
      float inv = 1.0f / sum;
#pragma unroll
      for (int j = 0; j < 4; ++j)
        s_p[(m_base + 8*g + r)*PV + j*16 + ln] = (h16)(s[j][r] * inv);
    }
    asm volatile("s_wait_dscnt 0" ::: "memory");  // P stores -> A reload (same wave)

    // O_h = P * V_h  (K-dim = 64 -> two k-steps; Vt gives contiguous B frags)
    v16h ap0 = loadA(s_p, PV, m_base, 0,  lane);
    v16h ap1 = loadA(s_p, PV, m_base, 32, lane);
#pragma unroll
    for (int dt = 0; dt < 2; ++dt) {
      v8f o = {};
      o = WMMA(ap0, loadB(s_vt, PV, h*32 + dt*16, 0,  lane), o);
      o = WMMA(ap1, loadB(s_vt, PV, h*32 + dt*16, 32, lane), o);
      int c = h*32 + dt*16 + ln;
#pragma unroll
      for (int r = 0; r < 8; ++r)
        s_o[(m_base + 8*g + r)*PX + c] = (h16)o[r];
    }
  }

  // ---- stage 4: projection + scatter to BCHW output (wave-private rows) ----
  v16h oa[4];
#pragma unroll
  for (int ks = 0; ks < 4; ++ks) oa[ks] = loadA(s_o, PX, m_base, 32*ks, lane);

#pragma unroll
  for (int nt = 0; nt < 8; ++nt) {
    v8f acc = {};
#pragma unroll
    for (int ks = 0; ks < 4; ++ks)
      acc = WMMA(oa[ks], loadB(WpT, 128, nt*16, 32*ks, lane), acc);
    int c = nt*16 + ln;
    float bias = bproj[c];
    int iy = (m_base + 8*g) >> 3;            // window-local row; ix == r
    int rowbase = ((b*128 + c)*256 + hy1*8 + iy)*256 + wx1*8;
    v4f lo, hi;
#pragma unroll
    for (int r = 0; r < 4; ++r) { lo[r] = acc[r] + bias; hi[r] = acc[r+4] + bias; }
    *(v4f*)(out + rowbase)     = lo;         // 8 consecutive w's per lane -> 2x b128
    *(v4f*)(out + rowbase + 4) = hi;
  }
}

extern "C" void kernel_launch(void* const* d_in, const int* in_sizes, int n_in,
                              void* d_out, int out_size, void* d_ws, size_t ws_size,
                              hipStream_t stream) {
  const float* x     = (const float*)d_in[0];
  const float* Wq    = (const float*)d_in[1];
  const float* bq    = (const float*)d_in[2];
  const float* Wkv   = (const float*)d_in[3];
  const float* bkv   = (const float*)d_in[4];
  const float* relb  = (const float*)d_in[5];
  const float* Wproj = (const float*)d_in[6];
  const float* bproj = (const float*)d_in[7];

  h16* WqT  = (h16*)d_ws;            // 128*128 f16
  h16* WkvT = WqT  + 128*128;        // 256*128 f16
  h16* WpT  = WkvT + 256*128;        // 128*128 f16  (total 128 KB of d_ws)

  prep_weights<<<128, 256, 0, stream>>>(Wq, Wkv, Wproj, WqT, WkvT, WpT);

  size_t smem = (size_t)(64*PX + 64*PX + 128*PV + 64*PV + 64*PX) * sizeof(h16); // 79872 B
  winattn_fused<<<4096, 128, smem, stream>>>(x, bq, bkv, relb, bproj,
                                             WqT, WkvT, WpT, (float*)d_out);
}
